// GRU_11562051961105
// MI455X (gfx1250) — matrix operand order repaired and checked by reading, not compiled, not run on hardware
//
#include <hip/hip_runtime.h>

// ---------------------------------------------------------------------------
// 2-layer GRU LM forward on gfx1250 using v_wmma_f32_16x16x32_bf16.
// Weights pre-converted (and pre-transposed to K-contiguous) into d_ws once
// per launch; recurrent state kept in fp32 + bf16 ping-pong buffers.
// Register-blocked GEMMs: GRU wave = 2 M-tiles x 6 gate GEMMs (12 WMMA/k-step,
// 10 fragment loads); FC wave = 4 M-tiles (4 WMMA/k-step, 5 fragment loads).
// ---------------------------------------------------------------------------

typedef __bf16 bf16_t;
typedef __attribute__((ext_vector_type(16))) __bf16 v16bf;
typedef __attribute__((ext_vector_type(8)))  __bf16 v8bf;
typedef __attribute__((ext_vector_type(8)))  float  v8f;

#define HID   1024
#define EMB   1024
#define VOCAB 10000
#define SEQ   64
#define BATCH 64

// ----------------------------- WMMA fragment loaders -----------------------
// A: 16x32 bf16 tile, row-major source with row stride HID.
// Layout: lanes 0-15 hold M=lane, K=0..15; lanes 16-31 hold
// M=lane-16, K=16..31.  (base points at X[mbase*HID + k0])
__device__ __forceinline__ v16bf load_a_frag(const bf16_t* __restrict__ base) {
  const int lane = threadIdx.x & 31;
  const int m    = lane & 15;
  const int kb   = (lane >> 4) << 4;          // 0 or 16
  return *reinterpret_cast<const v16bf*>(base + m * HID + kb);
}

// B: 32x16 bf16 tile. Wt is [N][K=HID] (K contiguous per output column).
// Contiguous K order, same as A: lane = column N, lanes 0-15 hold K=0..15, lanes 16-31 K=16..31.
__device__ __forceinline__ v16bf load_b_frag(const bf16_t* __restrict__ Wt,
                                             int nbase, int k0) {
  const int lane = threadIdx.x & 31;
  const int col  = lane & 15;
  const int kb   = (lane >> 4) << 4;          // 0 or 16
  return *reinterpret_cast<const v16bf*>(Wt + (size_t)(nbase + col) * HID + k0 + kb);
}

__device__ __forceinline__ v8f wmma_bf16(v16bf a, v16bf b, v8f c) {
  return __builtin_amdgcn_wmma_f32_16x16x32_bf16(false, a, false, b, (short)0, c,
                                                 false, false);
}

__device__ __forceinline__ float sigmoidf_(float x) {
  return 1.0f / (1.0f + __expf(-x));
}

// ----------------------------- setup kernels -------------------------------
// W [K=HID][N=3H] fp32  ->  Wt [3H][K=HID] bf16  (K-contiguous)
__global__ void transpose_convert_w(const float* __restrict__ W,
                                    bf16_t* __restrict__ Wt) {
  int idx = blockIdx.x * blockDim.x + threadIdx.x;   // over 3*HID*HID
  int n = idx >> 10;            // output row (0..3071)
  int k = idx & 1023;           // output col (0..1023)
  Wt[idx] = (bf16_t)W[(size_t)k * (3 * HID) + n];
}

// fcW [VOCAB][HID] fp32 -> bf16 (already K-contiguous for h1 @ fcW.T)
__global__ void convert_f32_bf16(const float* __restrict__ src,
                                 bf16_t* __restrict__ dst, int n) {
  int idx = blockIdx.x * blockDim.x + threadIdx.x;
  if (idx < n) dst[idx] = (bf16_t)src[idx];
}

// hidden [2][B][H] fp32 -> state buffers (fp32 + bf16), buffer slot 0
__global__ void init_hidden(const float* __restrict__ hidden,
                            float* __restrict__ h0f, float* __restrict__ h1f,
                            bf16_t* __restrict__ h0b, bf16_t* __restrict__ h1b) {
  int i = blockIdx.x * blockDim.x + threadIdx.x;
  if (i < BATCH * HID) {
    float a = hidden[i];
    float b = hidden[BATCH * HID + i];
    h0f[i] = a; h0b[i] = (bf16_t)a;
    h1f[i] = b; h1b[i] = (bf16_t)b;
  }
}

// x = bf16(emb[tok[b]])
__global__ void gather_x(const int* __restrict__ toks,
                         const float* __restrict__ emb,
                         bf16_t* __restrict__ xbf) {
  int b = blockIdx.x;
  int tok = toks[b];
  const float* src = emb + (size_t)tok * EMB;
  bf16_t* dst = xbf + b * EMB;
  for (int e = threadIdx.x; e < EMB; e += blockDim.x) dst[e] = (bf16_t)src[e];
}

__global__ void copy_hidden_out(const float* __restrict__ h0,
                                const float* __restrict__ h1,
                                float* __restrict__ dst) {
  int i = blockIdx.x * blockDim.x + threadIdx.x;
  if (i < BATCH * HID) {
    dst[i] = h0[i];
    dst[BATCH * HID + i] = h1[i];
  }
}

// ----------------------------- GRU cell kernel -----------------------------
// One wave computes a 32x16 tile of h_new (2 M-tiles), i.e. 12 accumulators
// (2 M-tiles x {gx,gh} x {r,z,n}).  Per K-step: 4 A frags + 6 B frags feed
// 12 WMMAs.  Block = 2 waves (covering batch 64); grid = 64 N tiles.
__global__ void __launch_bounds__(64) gru_cell_wmma(
    const bf16_t* __restrict__ xbf,       // [B][HID] layer input, bf16
    const bf16_t* __restrict__ hprev_bf,  // [B][HID]
    const float*  __restrict__ hprev_f,   // [B][HID]
    const bf16_t* __restrict__ Wt,        // [3H][HID] K-contiguous
    const bf16_t* __restrict__ Ut,        // [3H][HID]
    const float*  __restrict__ bw,        // [3H]
    const float*  __restrict__ bu,        // [3H]
    float*  __restrict__ hnew_f,          // [B][HID]
    bf16_t* __restrict__ hnew_bf)         // [B][HID]
{
  const int wid    = threadIdx.x >> 5;    // 0..1 -> rows [wid*32, wid*32+32)
  const int n_tile = blockIdx.x;          // 0..63
  const int m0     = wid * 32;
  const int nbase  = n_tile * 16;

  v8f accx[3][2];   // gx accumulators: [gate r/z/n][m-tile]
  v8f acch[3][2];   // gh accumulators
#pragma unroll
  for (int g = 0; g < 3; ++g)
#pragma unroll
    for (int t = 0; t < 2; ++t) { accx[g][t] = (v8f){}; acch[g][t] = (v8f){}; }

  for (int k0 = 0; k0 < HID; k0 += 32) {
    v16bf ax[2], ah[2], bwf[3], buf_[3];
#pragma unroll
    for (int t = 0; t < 2; ++t) {
      ax[t] = load_a_frag(xbf      + (size_t)(m0 + t * 16) * HID + k0);
      ah[t] = load_a_frag(hprev_bf + (size_t)(m0 + t * 16) * HID + k0);
    }
#pragma unroll
    for (int g = 0; g < 3; ++g) {
      bwf[g]  = load_b_frag(Wt, g * HID + nbase, k0);
      buf_[g] = load_b_frag(Ut, g * HID + nbase, k0);
    }
#pragma unroll
    for (int g = 0; g < 3; ++g)
#pragma unroll
      for (int t = 0; t < 2; ++t) {
        accx[g][t] = wmma_bf16(ax[t], bwf[g],  accx[g][t]);
        acch[g][t] = wmma_bf16(ah[t], buf_[g], acch[g][t]);
      }
  }

  // C/D layout: lanes 0-15 -> rows base+v, col nbase+lane;
  //             lanes 16-31 -> rows base+8+v, col nbase+lane-16.
  const int lane = threadIdx.x & 31;
  const int col  = nbase + (lane & 15);
  const float bwr = bw[col], bwz = bw[HID + col], bwn = bw[2 * HID + col];
  const float bur = bu[col], buz = bu[HID + col], bun = bu[2 * HID + col];

#pragma unroll
  for (int t = 0; t < 2; ++t) {
    const int rowb = m0 + t * 16 + ((lane >> 4) << 3);
#pragma unroll
    for (int v = 0; v < 8; ++v) {
      const int row = rowb + v;
      float ir = accx[0][t][v] + bwr, iz = accx[1][t][v] + bwz,
            in_ = accx[2][t][v] + bwn;
      float hr = acch[0][t][v] + bur, hz = acch[1][t][v] + buz,
            hn = acch[2][t][v] + bun;
      float r  = sigmoidf_(ir + hr);
      float z  = sigmoidf_(iz + hz);
      float ht = tanhf(in_ + r * hn);
      float hp = hprev_f[(size_t)row * HID + col];
      float hv = (1.0f - z) * hp + z * ht;
      hnew_f [(size_t)row * HID + col] = hv;
      hnew_bf[(size_t)row * HID + col] = (bf16_t)hv;
    }
  }
}

// ----------------------------- FC (logits) kernel --------------------------
// logits[B][VOCAB] = h1 @ fcW.T + fcb.  VOCAB = 625 * 16 exactly.
// One wave computes the full batch (4 M-tiles) for its N tile: per K-step
// 1 B frag + 4 A frags feed 4 WMMAs.  Block = 8 waves = 8 N tiles.
__global__ void __launch_bounds__(256) fc_wmma(
    const bf16_t* __restrict__ h1bf,      // [B][HID]
    const bf16_t* __restrict__ fcWb,      // [VOCAB][HID] K-contiguous
    const float*  __restrict__ fcb,       // [VOCAB]
    float* __restrict__ out)              // [B][VOCAB] (this timestep)
{
  const int wid    = threadIdx.x >> 5;            // 0..7
  const int n_tile = blockIdx.x * 8 + wid;        // 0..631
  if (n_tile >= VOCAB / 16) return;               // uniform per-wave guard
  const int nbase  = n_tile * 16;

  v8f acc[4];
#pragma unroll
  for (int m = 0; m < 4; ++m) acc[m] = (v8f){};

  for (int k0 = 0; k0 < HID; k0 += 32) {
    v16bf b = load_b_frag(fcWb, nbase, k0);
#pragma unroll
    for (int m = 0; m < 4; ++m) {
      v16bf a = load_a_frag(h1bf + (size_t)(m * 16) * HID + k0);
      acc[m] = wmma_bf16(a, b, acc[m]);
    }
  }

  const int lane = threadIdx.x & 31;
  const int col  = nbase + (lane & 15);
  const float bias = fcb[col];
#pragma unroll
  for (int m = 0; m < 4; ++m) {
    const int rowb = m * 16 + ((lane >> 4) << 3);
#pragma unroll
    for (int v = 0; v < 8; ++v)
      out[(size_t)(rowb + v) * VOCAB + col] = acc[m][v] + bias;
  }
}

// ----------------------------- launch --------------------------------------
extern "C" void kernel_launch(void* const* d_in, const int* in_sizes, int n_in,
                              void* d_out, int out_size, void* d_ws, size_t ws_size,
                              hipStream_t stream) {
  (void)in_sizes; (void)n_in; (void)out_size; (void)ws_size;

  const int*   inputs = (const int*)  d_in[0];   // [SEQ][BATCH]
  const float* hidden = (const float*)d_in[1];   // [2][BATCH][HID]
  const float* emb    = (const float*)d_in[2];   // [VOCAB][EMB]
  const float* W0     = (const float*)d_in[3];
  const float* U0     = (const float*)d_in[4];
  const float* bw0    = (const float*)d_in[5];
  const float* bu0    = (const float*)d_in[6];
  const float* W1     = (const float*)d_in[7];
  const float* U1     = (const float*)d_in[8];
  const float* bw1    = (const float*)d_in[9];
  const float* bu1    = (const float*)d_in[10];
  const float* fcW    = (const float*)d_in[11];
  const float* fcb    = (const float*)d_in[12];
  float* out = (float*)d_out;                    // logits [SEQ][B][V], then hidden [2][B][H]

  // ---- workspace carve-up (all offsets 256B aligned) ----
  char* ws = (char*)d_ws;
  const size_t W_ELEMS  = (size_t)3 * HID * HID;         // 3,145,728
  const size_t W_BYTES  = W_ELEMS * sizeof(bf16_t);      // 6,291,456
  const size_t FC_ELEMS = (size_t)VOCAB * HID;           // 10,240,000
  const size_t ST_F     = (size_t)BATCH * HID * sizeof(float);   // 262,144
  const size_t ST_B     = (size_t)BATCH * HID * sizeof(bf16_t);  // 131,072

  size_t off = 0;
  bf16_t* W0t  = (bf16_t*)(ws + off); off += W_BYTES;
  bf16_t* U0t  = (bf16_t*)(ws + off); off += W_BYTES;
  bf16_t* W1t  = (bf16_t*)(ws + off); off += W_BYTES;
  bf16_t* U1t  = (bf16_t*)(ws + off); off += W_BYTES;
  bf16_t* fcWb = (bf16_t*)(ws + off); off += FC_ELEMS * sizeof(bf16_t);
  bf16_t* xbf  = (bf16_t*)(ws + off); off += ST_B;
  float*  h0f[2]; float* h1f[2]; bf16_t* h0b[2]; bf16_t* h1b[2];
  for (int i = 0; i < 2; ++i) { h0f[i] = (float*)(ws + off); off += ST_F; }
  for (int i = 0; i < 2; ++i) { h1f[i] = (float*)(ws + off); off += ST_F; }
  for (int i = 0; i < 2; ++i) { h0b[i] = (bf16_t*)(ws + off); off += ST_B; }
  for (int i = 0; i < 2; ++i) { h1b[i] = (bf16_t*)(ws + off); off += ST_B; }

  // ---- one-time (per launch) weight conversion ----
  {
    dim3 blk(256);
    dim3 grdW((unsigned)(W_ELEMS / 256));
    transpose_convert_w<<<grdW, blk, 0, stream>>>(W0, W0t);
    transpose_convert_w<<<grdW, blk, 0, stream>>>(U0, U0t);
    transpose_convert_w<<<grdW, blk, 0, stream>>>(W1, W1t);
    transpose_convert_w<<<grdW, blk, 0, stream>>>(U1, U1t);
    dim3 grdF((unsigned)((FC_ELEMS + 255) / 256));
    convert_f32_bf16<<<grdF, blk, 0, stream>>>(fcW, fcWb, (int)FC_ELEMS);
    init_hidden<<<dim3(BATCH * HID / 256), blk, 0, stream>>>(
        hidden, h0f[0], h1f[0], h0b[0], h1b[0]);
  }

  // ---- recurrent scan ----
  for (int t = 0; t < SEQ; ++t) {
    const int r = t & 1;        // read slot
    const int w = 1 - r;        // write slot

    gather_x<<<dim3(BATCH), dim3(256), 0, stream>>>(inputs + t * BATCH, emb, xbf);

    // layer 0: input x, state h0
    gru_cell_wmma<<<dim3(HID / 16), dim3(64), 0, stream>>>(
        xbf, h0b[r], h0f[r], W0t, U0t, bw0, bu0, h0f[w], h0b[w]);

    // layer 1: input h0_new, state h1
    gru_cell_wmma<<<dim3(HID / 16), dim3(64), 0, stream>>>(
        h0b[w], h1b[r], h1f[r], W1t, U1t, bw1, bu1, h1f[w], h1b[w]);

    // logits for this timestep
    fc_wmma<<<dim3((VOCAB / 16 + 7) / 8), dim3(256), 0, stream>>>(
        h1b[w], fcWb, fcb, out + (size_t)t * BATCH * VOCAB);
  }

  // final hidden states: after t = SEQ-1, write slot was 1 - (63 & 1) = 0
  copy_hidden_out<<<dim3(BATCH * HID / 256), dim3(256), 0, stream>>>(
      h0f[0], h1f[0], out + (size_t)SEQ * BATCH * VOCAB);
}
